// Model_29772713296149
// MI455X (gfx1250) — compile-verified, matrix operand order checked by reading, not run on hardware
//
#include <hip/hip_runtime.h>
#include <hip/hip_bf16.h>
#include <math.h>

#define PI_F 3.14159265358979323846f

constexpr int Bn = 32, Ln = 512, Hn = 336, Cn = 321, Kn = 8, Sn = 8;
constexpr int FREQn = Hn / 2 + 1;   // 169
constexpr int Gn    = 2 * Kn + 1;   // 17
constexpr int BCn   = Bn * Cn;      // 10272 (=642*16)
constexpr int FPn   = 176;          // FREQ padded to multiple of 16
constexpr int DKn   = 352;          // padded K-dim for DFT (336->352) and irfft (2*FP)
constexpr int ZLDn  = 544;          // 3*FP padded to multiple of 32
constexpr int FWn   = Ln + Hn;      // 848: full row = [xn(512) | y_hat(336)]
constexpr int CWAVES = 6;           // waves per combine block (642 = 6*107)

typedef _Float16 v16h __attribute__((ext_vector_type(16)));
typedef float    v8f  __attribute__((ext_vector_type(8)));

// A/B fragment load for v_wmma_f32_16x16x32_f16 (k = 8*hs+j, then 16+8*hs+j, j<8).
// Lane m = lane&15 selects the row (A) / column (B); hs = lane>>4 selects K half.
__device__ __forceinline__ v16h load_frag(const _Float16* p, int hs) {
  v16h v;
  const int o1 = hs * 8, o2 = 16 + hs * 8;
#pragma unroll
  for (int j = 0; j < 8; ++j) v[j] = p[o1 + j];
#pragma unroll
  for (int j = 0; j < 8; ++j) v[8 + j] = p[o2 + j];
  return v;
}

__device__ __forceinline__ v8f wmma16(v16h a, v16h b, v8f c) {
  return __builtin_amdgcn_wmma_f32_16x16x32_f16(false, a, false, b, (short)0, c,
                                                false, false);
}

// Async global -> LDS copy of one 16B chunk (gfx1250 GLOBAL_LOAD_ASYNC_TO_LDS_B128,
// ASYNCcnt-tracked; LDS offset = low 32 bits of generic shared address).
__device__ __forceinline__ void async_g2l_b128(void* lds_ptr, const void* gptr) {
  unsigned int laddr = (unsigned int)(unsigned long long)(uintptr_t)lds_ptr;
  unsigned long long gaddr = (unsigned long long)(uintptr_t)gptr;
  asm volatile("global_load_async_to_lds_b128 %0, %1, off"
               :: "v"(laddr), "v"(gaddr) : "memory");
}
__device__ __forceinline__ void wait_async0() {
  asm volatile("s_wait_asynccnt 0" ::: "memory");
}

// ---------------- table builders ----------------
__global__ void k_tab_wb(const float* __restrict__ Wb, _Float16* __restrict__ Wb_h) {
  int i = blockIdx.x * 256 + threadIdx.x;
  if (i < Hn * Ln) Wb_h[i] = (_Float16)Wb[i];
}

__global__ void k_tab_dft(_Float16* __restrict__ dre, _Float16* __restrict__ dim_) {
  int i = blockIdx.x * 256 + threadIdx.x;
  if (i >= FPn * DKn) return;
  int f = i / DKn, t = i % DKn;
  float re = 0.f, im = 0.f;
  if (f < FREQn && t < Hn) {
    float th = 2.f * PI_F * (float)f * (float)t / (float)Hn;
    re = cosf(th);
    im = -sinf(th);
  }
  dre[i] = (_Float16)re;
  dim_[i] = (_Float16)im;
}

__global__ void k_tab_idft(_Float16* __restrict__ idt) {
  int i = blockIdx.x * 256 + threadIdx.x;
  if (i >= Hn * DKn) return;
  int t = i / DKn, kp = i % DKn;
  int comp = kp / FPn, f = kp % FPn;
  float v = 0.f;
  if (f < FREQn) {
    float w = (f == 0 || f == FREQn - 1) ? 1.f : 2.f;
    float th = 2.f * PI_F * (float)f * (float)t / (float)Hn;
    v = (comp == 0 ? w * cosf(th) : -w * sinf(th)) / (float)Hn;
  }
  idt[i] = (_Float16)v;
}

__global__ void k_tab_wm(const float* __restrict__ Wr, const float* __restrict__ Wi,
                         _Float16* __restrict__ Whr, _Float16* __restrict__ Whi) {
  int i = blockIdx.x * 256 + threadIdx.x;
  if (i >= FPn * ZLDn) return;
  int g = i / ZLDn, kp = i % ZLDn;
  int comp = kp / FPn, f = kp % FPn;
  float vr = 0.f, vi = 0.f;
  if (g < FREQn && comp < 3 && f < FREQn) {
    int idx = g * (3 * FREQn) + comp * FREQn + f;
    vr = Wr[idx];
    vi = Wi[idx];
  }
  Whr[i] = (_Float16)vr;
  Whi[i] = (_Float16)vi;
}

// ---------------- stage 1: LayerNorm over L, write xn (f16) into full rows ----
__global__ __launch_bounds__(256) void k_layernorm(const float* __restrict__ x,
                                                   _Float16* __restrict__ full_h,
                                                   float* __restrict__ mu_o,
                                                   float* __restrict__ norm_o) {
  __shared__ float r1[256], r2[256];
  __shared__ float smu, sinv;
  int bc = blockIdx.x;
  int b = bc / Cn, c = bc % Cn;
  int tid = threadIdx.x;
  const float* xp = x + (size_t)b * Ln * Cn + c;
  float s1 = 0.f, s2 = 0.f;
  for (int l = tid; l < Ln; l += 256) {
    float v = xp[(size_t)l * Cn];
    s1 += v;
    s2 += v * v;
  }
  r1[tid] = s1;
  r2[tid] = s2;
  __syncthreads();
  for (int off = 128; off > 0; off >>= 1) {
    if (tid < off) {
      r1[tid] += r1[tid + off];
      r2[tid] += r2[tid + off];
    }
    __syncthreads();
  }
  if (tid == 0) {
    float mu = r1[0] / (float)Ln;
    float var = r2[0] / (float)Ln - mu * mu;
    float nr = sqrtf(var + 1e-5f);
    mu_o[bc] = mu;
    norm_o[bc] = nr;
    smu = mu;
    sinv = 1.f / nr;
  }
  __syncthreads();
  _Float16* out = full_h + (size_t)bc * FWn;
  for (int l = tid; l < Ln; l += 256)
    out[l] = (_Float16)((xp[(size_t)l * Cn] - smu) * sinv);
}

// ---------------- stage 2: y_hat = xn @ Wb^T + bb (WMMA GEMM) -----------------
__global__ __launch_bounds__(32) void k_gemm_yhat(_Float16* __restrict__ full_h,
                                                  const _Float16* __restrict__ Wb_h,
                                                  const float* __restrict__ bb,
                                                  float* __restrict__ yhat32) {
  int m0 = blockIdx.x * 16, n0 = blockIdx.y * 16;
  int lane = threadIdx.x, ml = lane & 15, hs = lane >> 4;
  const _Float16* arow = full_h + (size_t)(m0 + ml) * FWn;   // xn region
  const _Float16* brow = Wb_h + (size_t)(n0 + ml) * Ln;      // Wb[h][l] contiguous
  v8f acc = {};
  for (int kk = 0; kk < Ln; kk += 32)
    acc = wmma16(load_frag(arow + kk, hs), load_frag(brow + kk, hs), acc);
  int h = n0 + ml;
  float bias = bb[h];
#pragma unroll
  for (int r = 0; r < 8; ++r) {
    int row = m0 + 8 * hs + r;
    float v = acc[r] + bias;
    yhat32[(size_t)row * Hn + h] = v;
    full_h[(size_t)row * FWn + Ln + h] = (_Float16)v;
  }
}

// ---------------- stage 3: r_soft / feat / p = softmax(feat@Wc^T+bc) ----------
__global__ __launch_bounds__(32) void k_feat(const _Float16* __restrict__ full_h,
                                             const float* __restrict__ r,
                                             const float* __restrict__ Wc,
                                             const float* __restrict__ bcv,
                                             float* __restrict__ p_o) {
  int bc = blockIdx.x;
  int lane = threadIdx.x;
  float e[Kn];
  float mx = 1.f;
#pragma unroll
  for (int k = 0; k < Kn; ++k) {
    float a = fabsf(r[(size_t)bc * Kn + k]);
    e[k] = a;
    mx = fmaxf(mx, a);
  }
  float den = expf(1.f - mx);
#pragma unroll
  for (int k = 0; k < Kn; ++k) {
    e[k] = expf(e[k] - mx);
    den += e[k];
  }
  float rinv = 1.f / den;
  float acc[Sn];
#pragma unroll
  for (int s = 0; s < Sn; ++s) acc[s] = 0.f;
  const _Float16* xn = full_h + (size_t)bc * FWn;
  for (int l = lane; l < Ln; l += 32) {
    float v = (float)xn[l];
#pragma unroll
    for (int s = 0; s < Sn; ++s) acc[s] += v * Wc[s * (Ln + Kn) + l];
  }
  if (lane < Kn) {
    float v = e[lane] * rinv;
#pragma unroll
    for (int s = 0; s < Sn; ++s) acc[s] += v * Wc[s * (Ln + Kn) + Ln + lane];
  }
  for (int off = 16; off; off >>= 1)
#pragma unroll
    for (int s = 0; s < Sn; ++s) acc[s] += __shfl_xor(acc[s], off);
  float mx2 = -1e30f;
#pragma unroll
  for (int s = 0; s < Sn; ++s) {
    acc[s] += bcv[s];
    mx2 = fmaxf(mx2, acc[s]);
  }
  float d = 0.f;
#pragma unroll
  for (int s = 0; s < Sn; ++s) {
    acc[s] = expf(acc[s] - mx2);
    d += acc[s];
  }
  if (lane == 0) {
    float di = 1.f / d;
#pragma unroll
    for (int s = 0; s < Sn; ++s) p_o[(size_t)bc * Sn + s] = acc[s] * di;
  }
}

// ------- stage 4: fused gather + DFT (WMMA) + filter mul + k-reduction --------
// 6 waves / block share the states slice, p tile and the async-staged DFT tile.
__global__ __launch_bounds__(CWAVES * 32) void k_combine(
    const _Float16* __restrict__ full_h, const float* __restrict__ p,
    const int* __restrict__ leader, const int* __restrict__ shiftv,
    const float* __restrict__ st_re, const float* __restrict__ st_im,
    const _Float16* __restrict__ dre, const _Float16* __restrict__ dim_,
    _Float16* __restrict__ z_re, _Float16* __restrict__ z_im) {
  __shared__ float st[Sn * Gn * 16 * 2];          // states slice (17.4 KB)
  __shared__ float pl[CWAVES * 16 * Sn];          // p tile (3 KB)
  __shared__ _Float16 sdft[2 * 16 * DKn];         // [tab][f][t] DFT tile (22.5 KB)
  int tid = threadIdx.x;
  int wv = tid >> 5, lane = tid & 31;
  int ml = lane & 15, hs = lane >> 4;
  int mblk = blockIdx.x * (CWAVES * 16);          // first row of this block
  int m0 = mblk + wv * 16;                        // this wave's row tile
  int f0 = blockIdx.y * 16;

  // --- async global->LDS staging of the DFT tile (both tables) ---
  constexpr int CHUNKS = 2 * 16 * (DKn / 8);      // 16B chunks (1408)
  for (int i = tid; i < CHUNKS; i += CWAVES * 32) {
    int tab = i / (16 * (DKn / 8));
    int rem = i % (16 * (DKn / 8));
    int fl = rem / (DKn / 8), ch = rem % (DKn / 8);
    const _Float16* g = (tab ? dim_ : dre) + (size_t)(f0 + fl) * DKn + ch * 8;
    _Float16* l = sdft + (size_t)tab * 16 * DKn + (size_t)fl * DKn + ch * 8;
    async_g2l_b128(l, g);
  }
  // --- states slice + p tile (ordinary stores) ---
  for (int i = tid; i < Sn * Gn * 16; i += CWAVES * 32) {
    int fl = i & 15, sg = i >> 4;
    int f = f0 + fl;
    float re = 0.f, im = 0.f;
    if (f < FREQn) {
      re = st_re[(size_t)sg * FREQn + f];
      im = st_im[(size_t)sg * FREQn + f];
    }
    st[i * 2] = re;
    st[i * 2 + 1] = im;
  }
  for (int i = tid; i < CWAVES * 16 * Sn; i += CWAVES * 32)
    pl[i] = p[(size_t)(mblk + (i >> 3)) * Sn + (i & 7)];
  wait_async0();
  __syncthreads();

  int rowA = m0 + ml;
  const _Float16* bre = sdft + (size_t)ml * DKn;
  const _Float16* bim = sdft + (size_t)16 * DKn + (size_t)ml * DKn;

  // y_hat_f via DFT GEMM (K padded 336->352, table zero-padded)
  v8f yre = {}, yim = {};
  {
    const _Float16* arow = full_h + (size_t)rowA * FWn + Ln;
    for (int kk = 0; kk < DKn; kk += 32) {
      v16h a = load_frag(arow + kk, hs);
      yre = wmma16(a, load_frag(bre + kk, hs), yre);
      yim = wmma16(a, load_frag(bim + kk, hs), yim);
    }
  }

  // per-lane leader window bases (prefetched one k ahead)
  int b = rowA / Cn;
  const _Float16* srows[Kn];
#pragma unroll
  for (int k = 0; k < Kn; ++k) {
    int lead = leader[(size_t)rowA * Kn + k];
    int sh = shiftv[(size_t)rowA * Kn + k];
    srows[k] = full_h + ((size_t)(b * Cn + lead) * FWn + (Ln - sh));
  }
  __builtin_prefetch(srows[0], 0, 0);

  float z1r[8] = {}, z1i[8] = {}, z2r[8] = {}, z2i[8] = {};
  for (int k = 0; k < Kn; ++k) {
    if (k + 1 < Kn) __builtin_prefetch(srows[k + 1], 0, 0);
    v8f sre = {}, sim = {};
    for (int kk = 0; kk < DKn; kk += 32) {
      v16h a = load_frag(srows[k] + kk, hs);
      sre = wmma16(a, load_frag(bre + kk, hs), sre);
      sim = wmma16(a, load_frag(bim + kk, hs), sim);
    }
#pragma unroll
    for (int r = 0; r < 8; ++r) {
      int rl = wv * 16 + 8 * hs + r;
      float fr1 = 0.f, fi1 = 0.f, fr2 = 0.f, fi2 = 0.f;
#pragma unroll
      for (int s = 0; s < Sn; ++s) {
        float ps = pl[rl * Sn + s];
        int i1 = ((s * Gn + k) * 16 + ml) * 2;
        int i2 = ((s * Gn + Kn + k) * 16 + ml) * 2;
        fr1 += ps * st[i1];
        fi1 += ps * st[i1 + 1];
        fr2 += ps * st[i2];
        fi2 += ps * st[i2 + 1];
      }
      float ar = sre[r], ai = sim[r];
      z1r[r] += ar * fr1 - ai * fi1;
      z1i[r] += ar * fi1 + ai * fr1;
      float dr0 = ar - yre[r], di0 = ai - yim[r];
      z2r[r] += dr0 * fr2 - di0 * fi2;
      z2i[r] += dr0 * fi2 + di0 * fr2;
    }
  }

  int f = f0 + ml;
  if (f < FREQn) {
#pragma unroll
    for (int r = 0; r < 8; ++r) {
      int rl = wv * 16 + 8 * hs + r;
      float fr = 0.f, fi = 0.f;
#pragma unroll
      for (int s = 0; s < Sn; ++s) {
        float ps = pl[rl * Sn + s];
        int i3 = ((s * Gn + 2 * Kn) * 16 + ml) * 2;
        fr += ps * st[i3];
        fi += ps * st[i3 + 1];
      }
      float z3r = yre[r] * fr - yim[r] * fi;
      float z3i = yre[r] * fi + yim[r] * fr;
      size_t base = (size_t)(m0 + 8 * hs + r) * ZLDn;
      z_re[base + f] = (_Float16)z1r[r];
      z_re[base + FPn + f] = (_Float16)z2r[r];
      z_re[base + 2 * FPn + f] = (_Float16)z3r;
      z_im[base + f] = (_Float16)z1i[r];
      z_im[base + FPn + f] = (_Float16)z2i[r];
      z_im[base + 2 * FPn + f] = (_Float16)z3i;
    }
  }
}

// ---------------- stage 5: complex mix GEMM: mixed = z @ Wm^T + bm ------------
__global__ __launch_bounds__(32) void k_gemm_mix(
    const _Float16* __restrict__ z_re, const _Float16* __restrict__ z_im,
    const _Float16* __restrict__ Wr, const _Float16* __restrict__ Wi,
    const float* __restrict__ bmr, const float* __restrict__ bmi,
    _Float16* __restrict__ mixed_h) {
  int m0 = blockIdx.x * 16, g0 = blockIdx.y * 16;
  int lane = threadIdx.x, ml = lane & 15, hs = lane >> 4;
  const _Float16* arr = z_re + (size_t)(m0 + ml) * ZLDn;
  const _Float16* ari = z_im + (size_t)(m0 + ml) * ZLDn;
  const _Float16* brr = Wr + (size_t)(g0 + ml) * ZLDn;
  const _Float16* bri = Wi + (size_t)(g0 + ml) * ZLDn;
  v8f aRR = {}, aII = {}, aRI = {}, aIR = {};
  for (int kk = 0; kk < ZLDn; kk += 32) {
    v16h ar = load_frag(arr + kk, hs);
    v16h ai = load_frag(ari + kk, hs);
    v16h br = load_frag(brr + kk, hs);
    v16h bi = load_frag(bri + kk, hs);
    aRR = wmma16(ar, br, aRR);
    aII = wmma16(ai, bi, aII);
    aRI = wmma16(ar, bi, aRI);
    aIR = wmma16(ai, br, aIR);
  }
  int g = g0 + ml;
  if (g < FREQn) {
    float br_ = bmr[g], bi_ = bmi[g];
#pragma unroll
    for (int r = 0; r < 8; ++r) {
      int row = m0 + 8 * hs + r;
      mixed_h[(size_t)row * DKn + g] = (_Float16)(aRR[r] - aII[r] + br_);
      mixed_h[(size_t)row * DKn + FPn + g] = (_Float16)(aRI[r] + aIR[r] + bi_);
    }
  }
}

// -------- stage 6: irfft GEMM + epilogue (y = (y_hat+ir)*norm + mu, transpose)
__global__ __launch_bounds__(32) void k_gemm_irfft(
    const _Float16* __restrict__ mixed_h, const _Float16* __restrict__ idt,
    const float* __restrict__ yhat32, const float* __restrict__ mu,
    const float* __restrict__ nrm, float* __restrict__ out) {
  int m0 = blockIdx.x * 16, t0 = blockIdx.y * 16;
  int lane = threadIdx.x, ml = lane & 15, hs = lane >> 4;
  const _Float16* arow = mixed_h + (size_t)(m0 + ml) * DKn;
  const _Float16* brow = idt + (size_t)(t0 + ml) * DKn;
  v8f acc = {};
  for (int kk = 0; kk < DKn; kk += 32)
    acc = wmma16(load_frag(arow + kk, hs), load_frag(brow + kk, hs), acc);
  int t = t0 + ml;
#pragma unroll
  for (int r = 0; r < 8; ++r) {
    int row = m0 + 8 * hs + r;
    float y = acc[r] + yhat32[(size_t)row * Hn + t];
    float res = y * nrm[row] + mu[row];
    int bb_ = row / Cn, cc = row % Cn;
    out[(size_t)bb_ * Hn * Cn + (size_t)t * Cn + cc] = res;
  }
}

// ------------------------------- launcher -------------------------------------
extern "C" void kernel_launch(void* const* d_in, const int* in_sizes, int n_in,
                              void* d_out, int out_size, void* d_ws, size_t ws_size,
                              hipStream_t stream) {
  const float* x      = (const float*)d_in[0];
  const float* r      = (const float*)d_in[1];
  const float* Wb     = (const float*)d_in[2];
  const float* bb     = (const float*)d_in[3];
  const float* Wc     = (const float*)d_in[4];
  const float* bcv    = (const float*)d_in[5];
  const float* st_re  = (const float*)d_in[6];
  const float* st_im  = (const float*)d_in[7];
  const float* Wm_re  = (const float*)d_in[8];
  const float* Wm_im  = (const float*)d_in[9];
  const float* bm_re  = (const float*)d_in[10];
  const float* bm_im  = (const float*)d_in[11];
  const int*   leader = (const int*)d_in[12];
  const int*   shiftv = (const int*)d_in[13];
  float* out = (float*)d_out;

  char* w = (char*)d_ws;
  auto alloc = [&](size_t bytes) -> void* {
    void* pp = (void*)w;
    w += (bytes + 255) & ~(size_t)255;
    return pp;
  };
  size_t full_bytes  = ((size_t)BCn * FWn + 64) * 2;
  size_t z_bytes     = ((size_t)BCn * ZLDn + 64) * 2;
  size_t mixed_bytes = ((size_t)BCn * DKn + 64) * 2;

  _Float16* full_h  = (_Float16*)alloc(full_bytes);
  float*    yhat32  = (float*)alloc((size_t)BCn * Hn * 4);
  float*    mu      = (float*)alloc((size_t)BCn * 4);
  float*    nrm     = (float*)alloc((size_t)BCn * 4);
  float*    p       = (float*)alloc((size_t)BCn * Sn * 4);
  _Float16* Wb_h    = (_Float16*)alloc((size_t)Hn * Ln * 2);
  _Float16* dre     = (_Float16*)alloc((size_t)FPn * DKn * 2);
  _Float16* dim_    = (_Float16*)alloc((size_t)FPn * DKn * 2);
  _Float16* idt     = (_Float16*)alloc((size_t)Hn * DKn * 2);
  _Float16* Wm_hr   = (_Float16*)alloc((size_t)FPn * ZLDn * 2);
  _Float16* Wm_hi   = (_Float16*)alloc((size_t)FPn * ZLDn * 2);
  _Float16* z_re    = (_Float16*)alloc(z_bytes);
  _Float16* z_im    = (_Float16*)alloc(z_bytes);
  _Float16* mixed_h = (_Float16*)alloc(mixed_bytes);

  // zero buffers whose pad regions feed WMMA (avoid NaN*0 poisoning)
  hipMemsetAsync(full_h, 0, full_bytes, stream);
  hipMemsetAsync(z_re, 0, z_bytes, stream);
  hipMemsetAsync(z_im, 0, z_bytes, stream);
  hipMemsetAsync(mixed_h, 0, mixed_bytes, stream);

  k_tab_wb<<<(Hn * Ln + 255) / 256, 256, 0, stream>>>(Wb, Wb_h);
  k_tab_dft<<<(FPn * DKn + 255) / 256, 256, 0, stream>>>(dre, dim_);
  k_tab_idft<<<(Hn * DKn + 255) / 256, 256, 0, stream>>>(idt);
  k_tab_wm<<<(FPn * ZLDn + 255) / 256, 256, 0, stream>>>(Wm_re, Wm_im, Wm_hr, Wm_hi);

  k_layernorm<<<BCn, 256, 0, stream>>>(x, full_h, mu, nrm);
  k_gemm_yhat<<<dim3(BCn / 16, Hn / 16), 32, 0, stream>>>(full_h, Wb_h, bb, yhat32);
  k_feat<<<BCn, 32, 0, stream>>>(full_h, r, Wc, bcv, p);
  k_combine<<<dim3(BCn / (16 * CWAVES), FPn / 16), CWAVES * 32, 0, stream>>>(
      full_h, p, leader, shiftv, st_re, st_im, dre, dim_, z_re, z_im);
  k_gemm_mix<<<dim3(BCn / 16, FPn / 16), 32, 0, stream>>>(
      z_re, z_im, Wm_hr, Wm_hi, bm_re, bm_im, mixed_h);
  k_gemm_irfft<<<dim3(BCn / 16, Hn / 16), 32, 0, stream>>>(
      mixed_h, idt, yhat32, mu, nrm, out);
}
